// GNN_11957188952439
// MI455X (gfx1250) — matrix operand order repaired and checked by reading, not compiled, not run on hardware
//
#include <hip/hip_runtime.h>
#include <hip/hip_bf16.h>

// ---------------------------------------------------------------------------
// SAGE-LSTM GNN for MI455X (gfx1250): wave32 + v_wmma_f32_16x16x32_f16.
// One wave = one 32-node tile (two 16-row WMMA tiles -> 2x B-fragment reuse,
// halving LDS bytes per FLOP vs M=16), one relation. LSTM gate GEMMs run on
// WMMA with f16 A/B (weights staged in LDS, activations in registers), f32
// accumulate. 4 waves/block = 1 wave/SIMD, ~315KB LDS/WGP (CDNA5 only).
// ---------------------------------------------------------------------------

typedef __attribute__((ext_vector_type(16))) _Float16 v16h;
typedef __attribute__((ext_vector_type(8)))  float    v8f;

#define NN    40000
#define FF    128
#define DD    16
#define GG    64
#define LDW   136          // padded LDS row stride (halfs): 272B = bank step 4
#define TILES 1250         // NN / 32
#define SMEM_BYTES (2*512*LDW*2 + 512*4 + 4*32*LDW*2)   // 315392 B < 320 KB

__device__ __forceinline__ v8f wmma16(v16h a, v16h b, v8f c) {
  return __builtin_amdgcn_wmma_f32_16x16x32_f16(
      /*neg_a=*/false, a, /*neg_b=*/false, b,
      /*c_mod=*/(short)0, c, /*reuse_a=*/false, /*reuse_b=*/false);
}

__device__ __forceinline__ float sigf(float x) {
  return 1.0f / (1.0f + __expf(-x));
}

__device__ __forceinline__ float tanh_fast(float x) {
#if __has_builtin(__builtin_amdgcn_tanhf)
  return __builtin_amdgcn_tanhf(x);
#else
  float e = __expf(2.0f * x);
  return 1.0f - 2.0f / (e + 1.0f);
#endif
}

// two 8-half LDS reads (p[0..7], p[16..23]) as one WMMA B fragment (2x ds_load_b128)
__device__ __forceinline__ v16h lds_loadB(const _Float16* p) {
  union { v16h v; uint4 q[2]; } u;
  const uint4* s = (const uint4*)p;
  u.q[0] = s[0];
  u.q[1] = s[2];
  return u.v;
}

// float atomic max via integer atomics (correct for any non-NaN floats)
__device__ __forceinline__ void atomicMaxF(float* a, float v) {
  if (v >= 0.0f) atomicMax((int*)a, __float_as_int(v));
  else           atomicMin((unsigned int*)a, __float_as_uint(v));
}

// Build a 16x32 f16 A-fragment row slice for this lane from a global f32 row.
// A layout (wave32): lanes 0-15 row=lane, K elems {k*32+0..7, k*32+16..23};
// lanes 16-31 row=lane-16, K elems {k*32+8..15, k*32+24..31}.
__device__ __forceinline__ v16h build_afrag(const float* frow, int k, int hi) {
  const float4* p0 = (const float4*)(frow + k * 32 + hi * 8);
  const float4* p1 = (const float4*)(frow + k * 32 + 16 + hi * 8);
  float4 a0 = p0[0], a1 = p0[1], b0 = p1[0], b1 = p1[1];
  v16h v;
  v[0]=(_Float16)a0.x;  v[1]=(_Float16)a0.y;  v[2]=(_Float16)a0.z;  v[3]=(_Float16)a0.w;
  v[4]=(_Float16)a1.x;  v[5]=(_Float16)a1.y;  v[6]=(_Float16)a1.z;  v[7]=(_Float16)a1.w;
  v[8]=(_Float16)b0.x;  v[9]=(_Float16)b0.y;  v[10]=(_Float16)b0.z; v[11]=(_Float16)b0.w;
  v[12]=(_Float16)b1.x; v[13]=(_Float16)b1.y; v[14]=(_Float16)b1.z; v[15]=(_Float16)b1.w;
  return v;
}

// ---------------------------------------------------------------------------
__global__ void __launch_bounds__(128, 1)
sage_lstm_kernel(const float* __restrict__ feat,
                 const int* __restrict__ nbr0, const int* __restrict__ deg0,
                 const int* __restrict__ nbr1, const int* __restrict__ deg1,
                 const int* __restrict__ nbr2, const int* __restrict__ deg2,
                 const int* __restrict__ nbr3, const int* __restrict__ deg3,
                 const float* __restrict__ Wih, const float* __restrict__ Whh,
                 const float* __restrict__ bias,
                 const float* __restrict__ Wself, const float* __restrict__ Wneigh,
                 const float* __restrict__ bout,
                 float* __restrict__ outA, float* __restrict__ outB)
{
  extern __shared__ __align__(16) unsigned char smem[];
  _Float16* sWih = (_Float16*)smem;                       // 512 x LDW halfs
  _Float16* sWhh = sWih + 512 * LDW;                      // 512 x LDW halfs
  float*    sB   = (float*)(sWhh + 512 * LDW);            // 512 f32
  _Float16* sH   = (_Float16*)(sB + 512);                 // 4 waves x 32 x LDW

  const int r    = blockIdx.x;
  const int tid  = threadIdx.x;
  const int wave = tid >> 5;
  const int lane = tid & 31;

  const int* nbr = (r == 0) ? nbr0 : (r == 1) ? nbr1 : (r == 2) ? nbr2 : nbr3;
  const int* deg = (r == 0) ? deg0 : (r == 1) ? deg1 : (r == 2) ? deg2 : deg3;
  float* outT = (r < 2) ? outA : outB;

  // Stage this relation's LSTM weights into LDS as f16 (padded rows).
  const float* gWih = Wih + (long)r * 512 * FF;
  const float* gWhh = Whh + (long)r * 512 * FF;
  const float* gB   = bias + r * 512;
  for (int idx = tid; idx < 512 * FF; idx += 128) {
    int row = idx >> 7, col = idx & 127;
    sWih[row * LDW + col] = (_Float16)gWih[idx];
    sWhh[row * LDW + col] = (_Float16)gWhh[idx];
  }
  for (int idx = tid; idx < 512; idx += 128) sB[idx] = gB[idx];
  __syncthreads();

  _Float16* myH = sH + wave * 32 * LDW;
  const int hi    = lane >> 4;     // 0: lanes 0-15, 1: lanes 16-31
  const int laneN = lane & 15;

  for (int tile = blockIdx.y * 4 + wave; tile < TILES; tile += gridDim.y * 4) {
    const int n0 = tile * 32;

    // deg for this lane's C-layout rows (tile m, VGPR e -> row m*16 + e + 8*hi)
    int degv[2][8];
#pragma unroll
    for (int m = 0; m < 2; ++m)
#pragma unroll
      for (int e = 0; e < 8; ++e) degv[m][e] = deg[n0 + m * 16 + e + hi * 8];

    // cell & hidden state in C layout: [m-tile][col-tile] x 8 rows
    v8f c_st[2][8], h_st[2][8];
#pragma unroll
    for (int m = 0; m < 2; ++m)
#pragma unroll
      for (int j = 0; j < 8; ++j)
#pragma unroll
        for (int e = 0; e < 8; ++e) { c_st[m][j][e] = 0.0f; h_st[m][j][e] = 0.0f; }

    // hidden-state A fragments (zero at t=0)
    v16h ah[2][4];
#pragma unroll
    for (int m = 0; m < 2; ++m)
#pragma unroll
      for (int k = 0; k < 4; ++k)
#pragma unroll
        for (int e = 0; e < 16; ++e) ah[m][k][e] = (_Float16)0.0f;

#pragma unroll 1
    for (int t = 0; t < DD; ++t) {
      // Gather x_t = feat[nbr[:, t]] directly into A fragments (f32 -> f16)
      const int nidA = nbr[(n0 + laneN) * DD + t];
      const int nidB = nbr[(n0 + 16 + laneN) * DD + t];
      const float* frowA = feat + (long)nidA * FF;
      const float* frowB = feat + (long)nidB * FF;
      v16h ax[2][4];
#pragma unroll
      for (int k = 0; k < 4; ++k) {
        ax[0][k] = build_afrag(frowA, k, hi);
        ax[1][k] = build_afrag(frowB, k, hi);
      }

      if (t + 1 < DD) {   // prefetch next timestep's gathered rows
        int p0 = nbr[(n0 + laneN) * DD + t + 1];
        int p1 = nbr[(n0 + 16 + laneN) * DD + t + 1];
        __builtin_prefetch(feat + (long)p0 * FF, 0, 0);
        __builtin_prefetch(feat + (long)p1 * FF, 0, 0);
      }

      // gates = x_t @ Wih^T + h @ Whh^T + b ; 16-col blocks of F, 4 gates each
#pragma unroll 1
      for (int cb = 0; cb < 8; ++cb) {
        const int ji = 0 * 8 + cb, jf = 1 * 8 + cb, jg = 2 * 8 + cb, jo = 3 * 8 + cb;
        float bi = sB[ji * 16 + laneN];
        float bf = sB[jf * 16 + laneN];
        float bg = sB[jg * 16 + laneN];
        float bq = sB[jo * 16 + laneN];
        v8f gi[2], gf[2], gg[2], go[2];
#pragma unroll
        for (int m = 0; m < 2; ++m)
#pragma unroll
          for (int e = 0; e < 8; ++e) {
            gi[m][e] = bi; gf[m][e] = bf; gg[m][e] = bg; go[m][e] = bq;
          }

#pragma unroll
        for (int k = 0; k < 4; ++k) {
          const int kb = k * 32 + hi * 8;    // B layout: lanes>=16 hold K 8..15, 24..31
          // 8 B fragments, each reused by both 16-row tiles (2x LDS reuse)
          v16h Bii = lds_loadB(sWih + (ji * 16 + laneN) * LDW + kb);
          v16h Bhi = lds_loadB(sWhh + (ji * 16 + laneN) * LDW + kb);
          v16h Bif = lds_loadB(sWih + (jf * 16 + laneN) * LDW + kb);
          v16h Bhf = lds_loadB(sWhh + (jf * 16 + laneN) * LDW + kb);
          v16h Big = lds_loadB(sWih + (jg * 16 + laneN) * LDW + kb);
          v16h Bhg = lds_loadB(sWhh + (jg * 16 + laneN) * LDW + kb);
          v16h Bio = lds_loadB(sWih + (jo * 16 + laneN) * LDW + kb);
          v16h Bho = lds_loadB(sWhh + (jo * 16 + laneN) * LDW + kb);
#pragma unroll
          for (int m = 0; m < 2; ++m) {
            gi[m] = wmma16(ax[m][k], Bii, gi[m]);
            gi[m] = wmma16(ah[m][k], Bhi, gi[m]);
            gf[m] = wmma16(ax[m][k], Bif, gf[m]);
            gf[m] = wmma16(ah[m][k], Bhf, gf[m]);
            gg[m] = wmma16(ax[m][k], Big, gg[m]);
            gg[m] = wmma16(ah[m][k], Bhg, gg[m]);
            go[m] = wmma16(ax[m][k], Bio, go[m]);
            go[m] = wmma16(ah[m][k], Bho, go[m]);
          }
        }

        // LSTM cell update with per-row validity mask (t < deg)
#pragma unroll
        for (int m = 0; m < 2; ++m)
#pragma unroll
          for (int e = 0; e < 8; ++e) {
            float iv = sigf(gi[m][e]);
            float fv = sigf(gf[m][e]);
            float gv = tanh_fast(gg[m][e]);
            float ov = sigf(go[m][e]);
            float cn = fv * c_st[m][cb][e] + iv * gv;
            float hn = ov * tanh_fast(cn);
            bool upd = (t < degv[m][e]);
            float cN = upd ? cn : c_st[m][cb][e];
            float hN = upd ? hn : h_st[m][cb][e];
            c_st[m][cb][e] = cN;
            h_st[m][cb][e] = hN;
            // spill h to per-wave LDS scratch (row-major, padded)
            myH[(m * 16 + e + hi * 8) * LDW + cb * 16 + laneN] = (_Float16)hN;
          }
      }

      // transpose h from C layout to A fragments for the next step
      // (DS ops from the same wave are in order: stores complete before loads)
#pragma unroll
      for (int m = 0; m < 2; ++m)
#pragma unroll
        for (int k = 0; k < 4; ++k) {
          const _Float16* pr = myH + (m * 16 + laneN) * LDW + k * 32 + hi * 8;
          union { v16h v; uint4 q[2]; } u;
          u.q[0] = *(const uint4*)pr;          // K group 0 (8 halfs)
          u.q[1] = *(const uint4*)(pr + 16);   // K group 1 (8 halfs)
          ah[m][k] = u.v;
        }
    } // t

    // out = feat_self @ Ws + h_fin @ Wn + b_out  (ah == h_fin fragments)
    v16h aself[2][4];
#pragma unroll
    for (int k = 0; k < 4; ++k) {
      aself[0][k] = build_afrag(feat + (long)(n0 + laneN) * FF, k, hi);
      aself[1][k] = build_afrag(feat + (long)(n0 + 16 + laneN) * FF, k, hi);
    }

    const float* WsR = Wself  + (long)r * FF * FF;
    const float* WnR = Wneigh + (long)r * FF * FF;
    const float* boR = bout   + r * FF;

#pragma unroll 1
    for (int j = 0; j < 8; ++j) {
      float bb = boR[j * 16 + laneN];
      v8f acc[2];
#pragma unroll
      for (int m = 0; m < 2; ++m)
#pragma unroll
        for (int e = 0; e < 8; ++e) acc[m][e] = bb;

#pragma unroll
      for (int k = 0; k < 4; ++k) {
        const int kb = k * 32 + hi * 8;
        v16h bs, bn;
#pragma unroll
        for (int q = 0; q < 16; ++q) {
          bs[q] = (_Float16)WsR[(long)(kb + q + (q >= 8 ? 8 : 0)) * FF + j * 16 + laneN];
          bn[q] = (_Float16)WnR[(long)(kb + q + (q >= 8 ? 8 : 0)) * FF + j * 16 + laneN];
        }
#pragma unroll
        for (int m = 0; m < 2; ++m) {
          acc[m] = wmma16(aself[m][k], bs, acc[m]);
          acc[m] = wmma16(ah[m][k],    bn, acc[m]);
        }
      }

      // pairwise max across relations via float atomic max
#pragma unroll
      for (int m = 0; m < 2; ++m)
#pragma unroll
        for (int e = 0; e < 8; ++e) {
          int row = n0 + m * 16 + e + hi * 8;
          atomicMaxF(outT + (long)row * FF + j * 16 + laneN, acc[m][e]);
        }
    }
  } // tile loop
}

// ---------------------------------------------------------------------------
// Small helper kernels: init, segment mean, final combine.
// ---------------------------------------------------------------------------
__global__ void init_kernel(float* outA, float* outB, float* segSums, float* segCnt) {
  long idx = (long)blockIdx.x * blockDim.x + threadIdx.x;
  const float ninf = -__builtin_inff();
  if (idx < (long)NN * FF) { outA[idx] = ninf; outB[idx] = ninf; }
  if (idx < GG * FF) segSums[idx] = 0.0f;
  if (idx < GG) segCnt[idx] = 0.0f;
}

__global__ void segsum_kernel(const float* __restrict__ feat,
                              const int* __restrict__ seg,
                              float* segSums, float* segCnt) {
  long idx = (long)blockIdx.x * blockDim.x + threadIdx.x;
  if (idx >= (long)NN * FF) return;
  int n = (int)(idx >> 7), c = (int)(idx & 127);
  int s = seg[n];
  atomicAdd(&segSums[s * FF + c], feat[idx]);
  if (c == 0) atomicAdd(&segCnt[s], 1.0f);
}

__global__ void mean_kernel(const float* segSums, const float* segCnt, float* mean) {
  int idx = blockIdx.x * blockDim.x + threadIdx.x;
  if (idx < GG * FF) mean[idx] = segSums[idx] / fmaxf(segCnt[idx >> 7], 1.0f);
}

__global__ void final_kernel(float* __restrict__ out, const float* __restrict__ pairB,
                             const float* __restrict__ mean, const int* __restrict__ seg) {
  long idx = (long)blockIdx.x * blockDim.x + threadIdx.x;
  if (idx >= (long)NN * FF) return;
  int n = (int)(idx >> 7), c = (int)(idx & 127);
  out[idx] = out[idx] + pairB[idx] + mean[seg[n] * FF + c];
}

// ---------------------------------------------------------------------------
extern "C" void kernel_launch(void* const* d_in, const int* in_sizes, int n_in,
                              void* d_out, int out_size, void* d_ws, size_t ws_size,
                              hipStream_t stream) {
  (void)in_sizes; (void)n_in; (void)out_size; (void)ws_size;
  const float* feat = (const float*)d_in[0];
  const int* nbr0 = (const int*)d_in[1]; const int* deg0 = (const int*)d_in[2];
  const int* nbr1 = (const int*)d_in[3]; const int* deg1 = (const int*)d_in[4];
  const int* nbr2 = (const int*)d_in[5]; const int* deg2 = (const int*)d_in[6];
  const int* nbr3 = (const int*)d_in[7]; const int* deg3 = (const int*)d_in[8];
  const int* seg  = (const int*)d_in[9];
  const float* Wih = (const float*)d_in[10];
  const float* Whh = (const float*)d_in[11];
  const float* b   = (const float*)d_in[12];
  const float* Ws  = (const float*)d_in[13];
  const float* Wn  = (const float*)d_in[14];
  const float* bo  = (const float*)d_in[15];

  float* outA    = (float*)d_out;                 // pair A: max(conv0, conv1)
  float* wsB     = (float*)d_ws;                  // pair B: max(conv2, conv3)
  float* segSums = wsB + (long)NN * FF;           // G x F
  float* segCnt  = segSums + GG * FF;             // G
  float* mean    = segCnt + GG;                   // G x F

  const long total = (long)NN * FF;
  const int  blk = 256;
  const int  g1 = (int)((total + blk - 1) / blk);

  init_kernel<<<g1, blk, 0, stream>>>(outA, wsB, segSums, segCnt);
  segsum_kernel<<<g1, blk, 0, stream>>>(feat, seg, segSums, segCnt);
  mean_kernel<<<(GG * FF + blk - 1) / blk, blk, 0, stream>>>(segSums, segCnt, mean);

  dim3 grid(4, (TILES + 3) / 4);   // (relation, node-tile blocks of 4 waves)
  sage_lstm_kernel<<<grid, 128, SMEM_BYTES, stream>>>(
      feat, nbr0, deg0, nbr1, deg1, nbr2, deg2, nbr3, deg3,
      Wih, Whh, b, Ws, Wn, bo, outA, wsB);

  final_kernel<<<g1, blk, 0, stream>>>(outA, wsB, mean, seg);
}
